// GAE_13073880449098
// MI455X (gfx1250) — matrix operand order repaired and checked by reading, not compiled, not run on hardware
//
#include <hip/hip_runtime.h>
#include <math.h>

// ---------------------------------------------------------------------------
// GAE with 4 GAT layers on gfx1250 (wave32, WMMA bf16).
// Heavy GEMMs (h_in@W and att@h) run on v_wmma_f32_16x16x32_bf16, register-
// blocked 2x4 tiles (32x64 output) per wave for A/B fragment reuse. One
// fragment set + compiler unroll: rounds 3/4 showed deeper manual pipelining
// spills to scratch at this tile size. att is materialized in bf16.
// ---------------------------------------------------------------------------

typedef __attribute__((ext_vector_type(16))) __bf16 v16bf;
typedef __attribute__((ext_vector_type(8)))  float  v8f;
typedef __attribute__((ext_vector_type(8)))  unsigned short us8;

#define GAT_N 4096
#define GAT_ALPHA 0.2f
#define GAT_NEGINF -9e15f

__device__ __forceinline__ unsigned short f2bf(float x) {
    unsigned int u = __float_as_uint(x);
    u += 0x7FFFu + ((u >> 16) & 1u);   // round-to-nearest-even
    return (unsigned short)(u >> 16);
}

// ------------------------- conversion kernels ------------------------------

// src f32 [M, ld_src] (logical width Kreal) -> dst bf16 [M, Kpad], zero padded
__global__ void gae_cvt_a_bf16(const float* __restrict__ src, int ld_src, int Kreal,
                               unsigned short* __restrict__ dst, int Kpad, long long total) {
    long long idx = (long long)blockIdx.x * blockDim.x + threadIdx.x;
    if (idx >= total) return;
    int k = (int)(idx % Kpad);
    long long m = idx / Kpad;
    float v = (k < Kreal) ? src[m * ld_src + k] : 0.0f;
    dst[idx] = f2bf(v);
}

// W f32 [Kreal, Nreal] row-major -> BT bf16 [Npad, Kpad] (BT[n,k] = W[k,n])
__global__ void gae_cvt_bt_bf16(const float* __restrict__ W, int Kreal, int Nreal,
                                unsigned short* __restrict__ BT, int Kpad, long long total) {
    long long idx = (long long)blockIdx.x * blockDim.x + threadIdx.x;
    if (idx >= total) return;
    int k = (int)(idx % Kpad);
    long long n = idx / Kpad;
    float v = (k < Kreal && n < Nreal) ? W[(long long)k * Nreal + n] : 0.0f;
    BT[idx] = f2bf(v);
}

// H f32 [M, ldh] (logical width Nreal) -> HT bf16 [Npad, M] (HT[n,m] = H[m,n])
__global__ void gae_cvt_ht_bf16(const float* __restrict__ H, int ldh, int Nreal,
                                unsigned short* __restrict__ HT, int Mrows, long long total) {
    long long idx = (long long)blockIdx.x * blockDim.x + threadIdx.x;
    if (idx >= total) return;
    int m = (int)(idx % Mrows);
    int n = (int)(idx / Mrows);
    float v = (n < Nreal) ? H[(long long)m * ldh + n] : 0.0f;
    HT[idx] = f2bf(v);
}

// ------------------------------ WMMA GEMM ----------------------------------
// C[M, ldc] (f32) = A(bf16, row-major [M, lda]) x BT(bf16, [N, ldb])^T
// Each wave computes a 32x64 output block = 2 (M) x 4 (N) WMMA tiles,
// reusing A fragments 4x and B fragments 2x. K stepped by 32.
// Requires: lda, ldb, K multiples of 32; M mult of 32; N mult of 64.
// EPI: 0 = none, 1 = ELU.
template <int EPI>
__global__ void gae_gemm_bf16_wmma(const unsigned short* __restrict__ A, int lda,
                                   const unsigned short* __restrict__ BT, int ldb,
                                   float* __restrict__ C, int ldc,
                                   int Mblk, int Nblk, int K) {
    int wave = threadIdx.x >> 5;
    int lane = threadIdx.x & 31;
    int blk  = blockIdx.x * (blockDim.x >> 5) + wave;
    if (blk >= Mblk * Nblk) return;
    int bm = (blk / Nblk) << 5;   // * 32
    int bn = (blk % Nblk) << 6;   // * 64
    int l15  = lane & 15;
    int half = lane >> 4;

    // ISA wave32 16-bit A layout: lane holds row bm+l15 (+16 for second tile);
    //   elems 0..7  -> K = k0 + 8*half .. +7
    //   elems 8..15 -> K = k0 + 16 + 8*half .. +7
    const unsigned short* arow0 = A + (long long)(bm + l15) * lda + 8 * half;
    const unsigned short* arow1 = arow0 + (long long)16 * lda;
    // ISA wave32 16-bit B layout: lane holds col bn+16t+l15;
    //   elems 0..7  -> K = k0 + 8*half .. +7
    //   elems 8..15 -> K = k0 + 16 + 8*half .. +7
    const unsigned short* brow[4];
#pragma unroll
    for (int t = 0; t < 4; ++t)
        brow[t] = BT + (long long)(bn + 16 * t + l15) * ldb + 8 * half;

    v8f acc[8];
#pragma unroll
    for (int q = 0; q < 8; ++q) acc[q] = (v8f){0.f,0.f,0.f,0.f,0.f,0.f,0.f,0.f};

    union Frag { v16bf v; us8 h[2]; };

    for (int k0 = 0; k0 < K; k0 += 32) {
        Frag fa0, fa1, fb[4];
        const us8* p;
        p = (const us8*)(arow0 + k0); fa0.h[0] = p[0]; fa0.h[1] = p[2];
        p = (const us8*)(arow1 + k0); fa1.h[0] = p[0]; fa1.h[1] = p[2];
#pragma unroll
        for (int t = 0; t < 4; ++t) {
            p = (const us8*)(brow[t] + k0);
            fb[t].h[0] = p[0]; fb[t].h[1] = p[2];
        }
#pragma unroll
        for (int t = 0; t < 4; ++t) {
            acc[t] = __builtin_amdgcn_wmma_f32_16x16x32_bf16(
                false, fa0.v, false, fb[t].v, (short)0, acc[t], false, false);
            acc[4 + t] = __builtin_amdgcn_wmma_f32_16x16x32_bf16(
                false, fa1.v, false, fb[t].v, (short)0, acc[4 + t], false, false);
        }
    }

    // C/D layout: VGPR r -> row +8*half + r, col +l15
#pragma unroll
    for (int mi = 0; mi < 2; ++mi) {
#pragma unroll
        for (int t = 0; t < 4; ++t) {
            float* crow = C + (long long)(bm + 16 * mi + 8 * half) * ldc + bn + 16 * t + l15;
            v8f a = acc[mi * 4 + t];
#pragma unroll
            for (int r = 0; r < 8; ++r) {
                float v = a[r];
                if (EPI == 1) v = (v > 0.0f) ? v : (__expf(v) - 1.0f);
                crow[(long long)r * ldc] = v;
            }
        }
    }
}

// --------------------------- small f32 kernels -----------------------------

// s[row] = dot(H[row,:F], a_self), n[row] = dot(H[row,:F], a_neigh); 1 wave/row
__global__ void gae_gemv2(const float* __restrict__ H, int ldh, int F,
                          const float* __restrict__ a_s, const float* __restrict__ a_n,
                          float* __restrict__ s, float* __restrict__ n, int Mrows) {
    int row  = blockIdx.x * (blockDim.x >> 5) + (threadIdx.x >> 5);
    int lane = threadIdx.x & 31;
    if (row >= Mrows) return;
    const float* hr = H + (long long)row * ldh;
    float accs = 0.0f, accn = 0.0f;
    for (int c = lane; c < F; c += 32) {
        float hv = hr[c];
        accs += hv * a_s[c];
        accn += hv * a_n[c];
    }
    for (int off = 16; off; off >>= 1) {
        accs += __shfl_down(accs, off);
        accn += __shfl_down(accn, off);
    }
    if (lane == 0) { s[row] = accs; n[row] = accn; }
}

// One block per row i: e = mask(leakyrelu((s_i + n_j) * M_ij)); softmax; bf16
__global__ void gae_att_rows(const float* __restrict__ Mm, const float* __restrict__ adj,
                             const float* __restrict__ s, const float* __restrict__ n,
                             unsigned short* __restrict__ att, int N) {
    __shared__ float ebuf[GAT_N];
    __shared__ float redm[8];
    __shared__ float reds[8];
    int i   = blockIdx.x;
    int tid = threadIdx.x;
    int wid = tid >> 5, lane = tid & 31;
    int nw  = blockDim.x >> 5;
    float si = s[i];
    const float* Mrow = Mm  + (long long)i * N;
    const float* Arow = adj + (long long)i * N;

    float lmax = -3.4e38f;
    for (int j = tid; j < N; j += blockDim.x) {
        float e = (si + n[j]) * Mrow[j];
        e = (e > 0.0f) ? e : GAT_ALPHA * e;
        e = (Arow[j] > 0.0f) ? e : GAT_NEGINF;
        ebuf[j] = e;
        lmax = fmaxf(lmax, e);
    }
    for (int off = 16; off; off >>= 1) lmax = fmaxf(lmax, __shfl_down(lmax, off));
    if (lane == 0) redm[wid] = lmax;
    __syncthreads();
    if (wid == 0) {
        float m = (lane < nw) ? redm[lane] : -3.4e38f;
        for (int off = 16; off; off >>= 1) m = fmaxf(m, __shfl_down(m, off));
        if (lane == 0) redm[0] = m;
    }
    __syncthreads();
    float gmax = redm[0];

    float lsum = 0.0f;
    for (int j = tid; j < N; j += blockDim.x) {
        float p = __expf(ebuf[j] - gmax);
        ebuf[j] = p;
        lsum += p;
    }
    for (int off = 16; off; off >>= 1) lsum += __shfl_down(lsum, off);
    if (lane == 0) reds[wid] = lsum;
    __syncthreads();
    if (wid == 0) {
        float t = (lane < nw) ? reds[lane] : 0.0f;
        for (int off = 16; off; off >>= 1) t += __shfl_down(t, off);
        if (lane == 0) reds[0] = t;
    }
    __syncthreads();
    float inv = 1.0f / reds[0];
    unsigned short* orow = att + (long long)i * N;
    for (int j = tid; j < N; j += blockDim.x) orow[j] = f2bf(ebuf[j] * inv);
}

// Row-L2 normalize Z[:, :F] -> out_r [M,F] and padded rbuf [M, ldr] (zeros pad)
__global__ void gae_norm_rows(const float* __restrict__ Z, int ldz, int F,
                              float* __restrict__ out_r, float* __restrict__ rbuf,
                              int ldr, int Mrows) {
    int row = blockIdx.x * blockDim.x + threadIdx.x;
    if (row >= Mrows) return;
    const float* zr = Z + (long long)row * ldz;
    float ss = 0.0f;
    for (int c = 0; c < F; ++c) { float v = zr[c]; ss += v * v; }
    float inv = 1.0f / fmaxf(sqrtf(ss), 1e-12f);
    for (int c = 0; c < F; ++c) {
        float v = zr[c] * inv;
        out_r[(long long)row * F + c] = v;
        rbuf[(long long)row * ldr + c] = v;
    }
    for (int c = F; c < ldr; ++c) rbuf[(long long)row * ldr + c] = 0.0f;
}

// A_pred[i,j] = sigmoid(dot(r_i, r_j)); ldr = 16, tail zero-padded
__global__ void gae_apred(const float* __restrict__ R, float* __restrict__ out, int N) {
    long long idx = (long long)blockIdx.x * blockDim.x + threadIdx.x;
    if (idx >= (long long)N * N) return;
    int i = (int)(idx / N);
    int j = (int)(idx % N);
    const float4* ri = (const float4*)(R + (long long)i * 16);
    const float4* rj = (const float4*)(R + (long long)j * 16);
    float d = 0.0f;
#pragma unroll
    for (int q = 0; q < 4; ++q) {
        float4 a = ri[q], b = rj[q];
        d += a.x * b.x + a.y * b.y + a.z * b.z + a.w * b.w;
    }
    out[idx] = 1.0f / (1.0f + __expf(-d));
}

// ------------------------------- launcher ----------------------------------

extern "C" void kernel_launch(void* const* d_in, const int* in_sizes, int n_in,
                              void* d_out, int out_size, void* d_ws, size_t ws_size,
                              hipStream_t stream) {
    (void)in_sizes; (void)n_in; (void)out_size; (void)ws_size;
    const int N = GAT_N;

    const float* x   = (const float*)d_in[0];
    const float* adj = (const float*)d_in[1];
    const float* Mm  = (const float*)d_in[2];
    const float* W[4], *As[4], *An[4];
    for (int l = 0; l < 4; ++l) {
        W[l]  = (const float*)d_in[3 + 3 * l];
        As[l] = (const float*)d_in[4 + 3 * l];
        An[l] = (const float*)d_in[5 + 3 * l];
    }

    // workspace layout (256B aligned slices)
    char* w = (char*)d_ws;
    auto alloc = [&](size_t bytes) {
        char* p = w;
        w += (bytes + 255) & ~(size_t)255;
        return p;
    };
    unsigned short* Abf = (unsigned short*)alloc((size_t)N * 2048 * 2);     // A operand bf16
    unsigned short* BTw = (unsigned short*)alloc((size_t)2048 * 2048 * 2);  // weight^T bf16
    unsigned short* HT  = (unsigned short*)alloc((size_t)2048 * N * 2);     // h^T bf16
    unsigned short* ATT = (unsigned short*)alloc((size_t)N * N * 2);        // softmax bf16
    float* Hb   = (float*)alloc((size_t)N * 2048 * 4);                      // h  f32
    float* Zb   = (float*)alloc((size_t)N * 2048 * 4);                      // att@h f32
    float* sv   = (float*)alloc((size_t)N * 4);
    float* nv   = (float*)alloc((size_t)N * 4);
    float* rbuf = (float*)alloc((size_t)N * 16 * 4);

    // Npad multiples of 64 (4 N-tiles per wave); Kpad multiples of 32.
    struct Layer { int fi, Kpad, fo, Npad; };
    const Layer L[4] = {
        {1433, 1440,  500,  512},
        { 500,  512,  500,  512},
        { 500,  512, 2000, 2048},
        {2000, 2016,   10,   64},
    };

    const float* cur = x;
    int ld_in = 1433;

    for (int l = 0; l < 4; ++l) {
        const int fi = L[l].fi, Kpad = L[l].Kpad, fo = L[l].fo, Npad = L[l].Npad;

        // 1. layer input -> bf16 A [N, Kpad]
        {
            long long tot = (long long)N * Kpad;
            gae_cvt_a_bf16<<<(int)((tot + 255) / 256), 256, 0, stream>>>(
                cur, ld_in, fi, Abf, Kpad, tot);
        }
        // 2. W -> bf16 B^T [Npad, Kpad]
        {
            long long tot = (long long)Npad * Kpad;
            gae_cvt_bt_bf16<<<(int)((tot + 255) / 256), 256, 0, stream>>>(
                W[l], fi, fo, BTw, Kpad, tot);
        }
        // 3. h = x @ W   (WMMA bf16, 32x64 blocks)
        {
            int blks = (N / 32) * (Npad / 64);
            gae_gemm_bf16_wmma<0><<<(blks + 7) / 8, 256, 0, stream>>>(
                Abf, Kpad, BTw, Kpad, Hb, Npad, N / 32, Npad / 64, Kpad);
        }
        // 4. s = h@a_self, n = h@a_neigh
        gae_gemv2<<<N / 8, 256, 0, stream>>>(Hb, Npad, fo, As[l], An[l], sv, nv, N);
        // 5. h -> bf16 h^T [Npad, N]  (B operand of att@h)
        {
            long long tot = (long long)Npad * N;
            gae_cvt_ht_bf16<<<(int)((tot + 255) / 256), 256, 0, stream>>>(
                Hb, Npad, fo, HT, N, tot);
        }
        // 6. attention rows: mask + leakyrelu + softmax -> bf16 att [N, N]
        gae_att_rows<<<N, 256, 0, stream>>>(Mm, adj, sv, nv, ATT, N);
        // 7. z = elu(att @ h)  (WMMA bf16, K = 4096, 32x64 blocks)
        {
            int blks = (N / 32) * (Npad / 64);
            gae_gemm_bf16_wmma<1><<<(blks + 7) / 8, 256, 0, stream>>>(
                ATT, N, HT, N, Zb, Npad, N / 32, Npad / 64, N);
        }
        cur = Zb;
        ld_in = Npad;
    }

    // finale: normalize r4 rows, emit r and A_pred = sigmoid(r r^T)
    float* out_A = (float*)d_out;
    float* out_r = out_A + (size_t)N * N;
    gae_norm_rows<<<(N + 255) / 256, 256, 0, stream>>>(Zb, 64, 10, out_r, rbuf, 16, N);
    {
        long long tot = (long long)N * N;
        gae_apred<<<(int)((tot + 255) / 256), 256, 0, stream>>>(rbuf, out_A, N);
    }
}
